// MultiHeadAttention_82506321756429
// MI455X (gfx1250) — matrix operand order repaired and checked by reading, not compiled, not run on hardware
//
#include <hip/hip_runtime.h>
#include <hip/hip_bf16.h>

// ---------------------------------------------------------------------------
// MultiHeadAttention on MI455X (gfx1250): f16 WMMA flash attention, wave32.
// Flash kernel uses 4-wave workgroups with LDS-shared K/V chunks to keep
// L2 traffic ~4x under the per-wave-streaming version.
// ---------------------------------------------------------------------------

typedef __attribute__((ext_vector_type(16))) _Float16 v16h;
typedef __attribute__((ext_vector_type(8)))  _Float16 v8h;
typedef __attribute__((ext_vector_type(8)))  float    v8f;

#define H_   8
#define IN_  256
#define HID_ 64
#define OUT_ 256
#define S_   4096
#define QW_  4   // waves (16-query tiles) per flash workgroup

static __device__ __forceinline__ v8f wmma_f16(v16h a, v16h b, v8f c) {
    // D = A(16x32 f16) * B(32x16 f16) + C(16x16 f32)
    return __builtin_amdgcn_wmma_f32_16x16x32_f16(
        /*neg_a=*/false, a, /*neg_b=*/false, b,
        /*c_mod=*/(short)0, c, /*reuse_a=*/false, /*reuse_b=*/false);
}

// reduction across a 16-lane group (lanes 0-15 or 16-31): xor masks < 16 stay
// inside the group on wave32.
static __device__ __forceinline__ float red_max16(float v) {
#pragma unroll
    for (int mask = 1; mask < 16; mask <<= 1)
        v = fmaxf(v, __shfl_xor(v, mask, 32));
    return v;
}
static __device__ __forceinline__ float red_sum16(float v) {
#pragma unroll
    for (int mask = 1; mask < 16; mask <<= 1)
        v += __shfl_xor(v, mask, 32);
    return v;
}

// ---------------------------------------------------------------------------
// K0: transpose + f32->f16 weights.
//   Wq/Wk/Wv [H][IN][HID] -> W*t [H][HID][IN]  (B-operand friendly)
//   Wo [H*HID][OUT]       -> Wot [OUT][H*HID]
// ---------------------------------------------------------------------------
__global__ void prep_weights(const float* __restrict__ Wq, const float* __restrict__ Wk,
                             const float* __restrict__ Wv, const float* __restrict__ Wo,
                             _Float16* __restrict__ Wqt, _Float16* __restrict__ Wkt,
                             _Float16* __restrict__ Wvt, _Float16* __restrict__ Wot) {
    const int stride = gridDim.x * blockDim.x;
    const int idx = blockIdx.x * blockDim.x + threadIdx.x;
    const int nW = H_ * IN_ * HID_;
    for (int t = idx; t < nW; t += stride) {
        int h = t / (IN_ * HID_);
        int r = t - h * (IN_ * HID_);
        int i = r / HID_;
        int d = r - i * HID_;
        int dst = (h * HID_ + d) * IN_ + i;
        Wqt[dst] = (_Float16)Wq[t];
        Wkt[dst] = (_Float16)Wk[t];
        Wvt[dst] = (_Float16)Wv[t];
    }
    const int nO = (H_ * HID_) * OUT_;
    for (int t = idx; t < nO; t += stride) {
        int c = t / OUT_;
        int o = t - c * OUT_;
        Wot[o * (H_ * HID_) + c] = (_Float16)Wo[t];
    }
}

// ---------------------------------------------------------------------------
// K1: per-head projections with WMMA. One wave per (16-row tile, head, matrix).
//   Q/K out: [H][S][HID] f16 (row-major).  V out: [H][HID][S] f16 (transposed).
// ---------------------------------------------------------------------------
__global__ __launch_bounds__(32) void proj_kernel(
    const float* __restrict__ q, const float* __restrict__ k, const float* __restrict__ v,
    const _Float16* __restrict__ Wqt, const _Float16* __restrict__ Wkt,
    const _Float16* __restrict__ Wvt,
    const float* __restrict__ bq, const float* __restrict__ bk, const float* __restrict__ bv,
    _Float16* __restrict__ Qp, _Float16* __restrict__ Kp, _Float16* __restrict__ Vt) {
    const int lane = threadIdx.x;
    const int s0 = blockIdx.x * 16;
    const int h = blockIdx.y;
    const int mat = blockIdx.z;  // 0=q, 1=k, 2=v

    const float* X = (mat == 0) ? q : (mat == 1) ? k : v;
    const _Float16* Wt = (mat == 0) ? Wqt : (mat == 1) ? Wkt : Wvt;
    const float* Bv = (mat == 0) ? bq : (mat == 1) ? bk : bv;

    const int m = lane & 15;   // row (A) / column N (B/C/D)
    const int hb = lane >> 4;  // half-wave
    const int kb = hb ? 16 : 0; // A-layout K base for this half

    v8f acc[4];
#pragma unroll
    for (int n = 0; n < 4; ++n) acc[n] = v8f{};

    for (int kc = 0; kc < IN_ / 32; ++kc) {
        // A tile 16x32: convert f32 -> f16. Contiguous k order used here: lanes0-15 K{0-15},
        // lanes16-31 K{16-31}. Two contiguous 8-float runs per lane.
        const float* xr = X + (size_t)(s0 + m) * IN_ + kc * 32;
        v8f x0 = *(const v8f*)(xr + kb);
        v8f x1 = *(const v8f*)(xr + kb + 8);
        v16h a;
#pragma unroll
        for (int e = 0; e < 8; ++e) {
            a[e] = (_Float16)x0[e];
            a[8 + e] = (_Float16)x1[e];
        }
#pragma unroll
        for (int n = 0; n < 4; ++n) {
            // B tile 32x16: lane = column (d), half = K-half; contiguous 16 f16.
            const _Float16* wr =
                Wt + ((size_t)h * HID_ + n * 16 + m) * IN_ + kc * 32 + hb * 16;
            v16h b = *(const v16h*)wr;
            acc[n] = wmma_f16(a, b, acc[n]);
        }
    }

#pragma unroll
    for (int n = 0; n < 4; ++n) {
        const int d = n * 16 + m;
        const float bias = Bv[h * HID_ + d];
#pragma unroll
        for (int r = 0; r < 8; ++r) {
            float val = acc[n][r] + bias;          // C/D: row = r + 8*hb, col = d
            const int srow = s0 + r + 8 * hb;
            if (mat == 2) {
                Vt[((size_t)h * HID_ + d) * S_ + srow] = (_Float16)val;   // transposed
            } else {
                _Float16* dst = (mat == 0) ? Qp : Kp;
                dst[((size_t)h * S_ + srow) * HID_ + d] = (_Float16)val;  // row-major
            }
        }
    }
}

// ---------------------------------------------------------------------------
// K2: flash attention. 4-wave workgroup = 64 query rows of one head.
//     Keys processed in chunks of 32; each chunk's K (32x64) and V (64x32)
//     staged once into LDS and consumed by all 4 waves.
//   Ocat out: [S][H*HID] f16 (heads concatenated, ready for output GEMM).
// ---------------------------------------------------------------------------
__global__ __launch_bounds__(32 * QW_) void flash_kernel(
    const _Float16* __restrict__ Qp, const _Float16* __restrict__ Kp,
    const _Float16* __restrict__ Vt, _Float16* __restrict__ Ocat) {
    __shared__ __align__(16) _Float16 Kls[32 * HID_];        // [key 0..31][d 0..63]
    __shared__ __align__(16) _Float16 Vls[HID_ * 32];        // [d 0..63][key 0..31]
    __shared__ __align__(16) _Float16 pstage[QW_][16 * 32];  // per-wave P staging

    const int tid = threadIdx.x;
    const int wave = tid >> 5;
    const int lane = tid & 31;
    const int s0 = blockIdx.x * (16 * QW_) + wave * 16;
    const int h = blockIdx.y;
    const int m = lane & 15;
    const int hb = lane >> 4;
    const int kb = hb ? 16 : 0;
    const float scale = 0.125f;  // 1/sqrt(HID)

    const _Float16* Krow = Kp + (size_t)h * S_ * HID_;
    const _Float16* Vrow = Vt + (size_t)h * HID_ * S_;

    // staging assignments (128 threads move 4KB + 4KB per chunk, 32B each)
    const int krow = tid >> 2, kpart = tid & 3;   // K: 32 rows x 4 x 16 f16
    const int vrow = tid >> 1, vpart = tid & 1;   // V: 64 rows x 2 x 16 f16
    const _Float16* kstage_src = Krow + (size_t)krow * HID_ + kpart * 16;
    const _Float16* vstage_src = Vrow + (size_t)vrow * S_ + vpart * 16;
    _Float16* kstage_dst = Kls + krow * HID_ + kpart * 16;
    _Float16* vstage_dst = Vls + vrow * 32 + vpart * 16;

    // Q tile A-operands for d=[0,32) and d=[32,64), loaded once.
    const _Float16* qr = Qp + ((size_t)h * S_ + s0 + m) * HID_;
    v16h aQ0, aQ1;
    {
        v8h t0 = *(const v8h*)(qr + kb);
        v8h t1 = *(const v8h*)(qr + kb + 8);
        v8h t2 = *(const v8h*)(qr + 32 + kb);
        v8h t3 = *(const v8h*)(qr + 32 + kb + 8);
#pragma unroll
        for (int e = 0; e < 8; ++e) {
            aQ0[e] = t0[e]; aQ0[8 + e] = t1[e];
            aQ1[e] = t2[e]; aQ1[8 + e] = t3[e];
        }
    }

    v8f O[4];
#pragma unroll
    for (int n = 0; n < 4; ++n) O[n] = v8f{};
    float mrow[8], lrow[8];
#pragma unroll
    for (int r = 0; r < 8; ++r) { mrow[r] = -1e30f; lrow[r] = 0.0f; }

    for (int c = 0; c < S_ / 32; ++c) {
        const int key0 = c * 32;

        // ---- cooperative staging of this chunk's K and V into LDS ----
        *(v16h*)kstage_dst = *(const v16h*)(kstage_src + (size_t)key0 * HID_);
        *(v16h*)vstage_dst = *(const v16h*)(vstage_src + key0);
        if (c + 1 < S_ / 32) {  // -> global_prefetch_b8 for next chunk
            __builtin_prefetch(kstage_src + (size_t)(key0 + 32) * HID_, 0, 3);
            __builtin_prefetch(vstage_src + key0 + 32, 0, 3);
        }
        __syncthreads();

        // ---- scores: two 16x16 tiles, K-contraction = 64 (2 WMMAs each) ----
        v8f sc[2];
#pragma unroll
        for (int t = 0; t < 2; ++t) {
            const _Float16* kr = Kls + (t * 16 + m) * HID_ + hb * 16;
            v16h b0 = *(const v16h*)(kr);        // d = hb*16 .. +15
            v16h b1 = *(const v16h*)(kr + 32);   // d = 32 + hb*16 .. +15
            v8f s = v8f{};
            s = wmma_f16(aQ0, b0, s);
            s = wmma_f16(aQ1, b1, s);
            sc[t] = s;
        }

        // ---- online softmax over the 32-key chunk ----
        float alpha[8];
#pragma unroll
        for (int r = 0; r < 8; ++r) {
            float v0 = sc[0][r] * scale;
            float v1 = sc[1][r] * scale;
            float mx = red_max16(fmaxf(v0, v1));
            float mn = fmaxf(mrow[r], mx);
            alpha[r] = __expf(mrow[r] - mn);
            float p0 = __expf(v0 - mn);
            float p1 = __expf(v1 - mn);
            sc[0][r] = p0;
            sc[1][r] = p1;
            lrow[r] = lrow[r] * alpha[r] + red_sum16(p0 + p1);
            mrow[r] = mn;
        }
#pragma unroll
        for (int n = 0; n < 4; ++n)
#pragma unroll
            for (int r = 0; r < 8; ++r) O[n][r] *= alpha[r];

        // ---- stage P (C/D layout) into this wave's LDS region, row-major ----
        _Float16* ps = pstage[wave];
#pragma unroll
        for (int t = 0; t < 2; ++t)
#pragma unroll
            for (int r = 0; r < 8; ++r)
                ps[(r + 8 * hb) * 32 + t * 16 + m] = (_Float16)sc[t][r];
        // same-wave DS ops are in-order; no cross-wave sharing of pstage.

        // ---- reload P in A layout ----
        v16h aP;
        {
            v8h p0 = *(const v8h*)(ps + m * 32 + kb);
            v8h p1 = *(const v8h*)(ps + m * 32 + kb + 8);
#pragma unroll
            for (int e = 0; e < 8; ++e) { aP[e] = p0[e]; aP[8 + e] = p1[e]; }
        }

        // ---- O += P @ V  (V chunk in LDS, row = d, contiguous keys) ----
#pragma unroll
        for (int n = 0; n < 4; ++n) {
            const _Float16* vr = Vls + (n * 16 + m) * 32 + hb * 16;
            v16h bv = *(const v16h*)vr;
            O[n] = wmma_f16(aP, bv, O[n]);
        }

        __syncthreads();  // WAR: all waves done with Kls/Vls before restaging
    }

    // ---- normalize + write concatenated heads ----
#pragma unroll
    for (int r = 0; r < 8; ++r) {
        const float inv = 1.0f / lrow[r];
        const int srow = s0 + r + 8 * hb;
#pragma unroll
        for (int n = 0; n < 4; ++n)
            Ocat[(size_t)srow * (H_ * HID_) + h * HID_ + n * 16 + m] =
                (_Float16)(O[n][r] * inv);
    }
}

// ---------------------------------------------------------------------------
// K3: output projection  out[S][256] = Ocat[S][512] @ Wo + bo   (f32 out)
// ---------------------------------------------------------------------------
__global__ __launch_bounds__(32) void outproj_kernel(
    const _Float16* __restrict__ Ocat, const _Float16* __restrict__ Wot,
    const float* __restrict__ bo, float* __restrict__ out) {
    const int lane = threadIdx.x;
    const int s0 = blockIdx.x * 16;
    const int o0 = blockIdx.y * 64;
    const int m = lane & 15;
    const int hb = lane >> 4;
    const int kb = hb ? 16 : 0;
    const int C = H_ * HID_;  // 512

    v8f acc[4];
#pragma unroll
    for (int n = 0; n < 4; ++n) acc[n] = v8f{};

    for (int kc = 0; kc < C / 32; ++kc) {
        const _Float16* ar = Ocat + (size_t)(s0 + m) * C + kc * 32;
        v8h t0 = *(const v8h*)(ar + kb);
        v8h t1 = *(const v8h*)(ar + kb + 8);
        v16h a;
#pragma unroll
        for (int e = 0; e < 8; ++e) { a[e] = t0[e]; a[8 + e] = t1[e]; }
#pragma unroll
        for (int n = 0; n < 4; ++n) {
            const _Float16* wr =
                Wot + (size_t)(o0 + n * 16 + m) * C + kc * 32 + hb * 16;
            v16h b = *(const v16h*)wr;
            acc[n] = wmma_f16(a, b, acc[n]);
        }
    }

#pragma unroll
    for (int n = 0; n < 4; ++n) {
        const int o = o0 + n * 16 + m;
        const float bias = bo[o];
#pragma unroll
        for (int r = 0; r < 8; ++r)
            out[(size_t)(s0 + r + 8 * hb) * OUT_ + o] = acc[n][r] + bias;
    }
}

// ---------------------------------------------------------------------------
// launch
// ---------------------------------------------------------------------------
extern "C" void kernel_launch(void* const* d_in, const int* in_sizes, int n_in,
                              void* d_out, int out_size, void* d_ws, size_t ws_size,
                              hipStream_t stream) {
    (void)in_sizes; (void)n_in; (void)out_size; (void)ws_size;

    const float* q  = (const float*)d_in[0];
    const float* k  = (const float*)d_in[1];
    const float* v  = (const float*)d_in[2];
    const float* Wq = (const float*)d_in[3];
    const float* bq = (const float*)d_in[4];
    const float* Wk = (const float*)d_in[5];
    const float* bk = (const float*)d_in[6];
    const float* Wv = (const float*)d_in[7];
    const float* bv = (const float*)d_in[8];
    const float* Wo = (const float*)d_in[9];
    const float* bo = (const float*)d_in[10];
    float* out = (float*)d_out;

    // workspace carve (256B aligned sub-buffers)
    char* ws = (char*)d_ws;
    const size_t WT_SZ = (size_t)H_ * HID_ * IN_ * sizeof(_Float16);      // 256 KB
    const size_t WOT_SZ = (size_t)OUT_ * H_ * HID_ * sizeof(_Float16);    // 256 KB
    const size_t P_SZ = (size_t)H_ * S_ * HID_ * sizeof(_Float16);        // 4 MB
    const size_t OC_SZ = (size_t)S_ * H_ * HID_ * sizeof(_Float16);       // 4 MB
    size_t off = 0;
    _Float16* Wqt = (_Float16*)(ws + off); off += WT_SZ;
    _Float16* Wkt = (_Float16*)(ws + off); off += WT_SZ;
    _Float16* Wvt = (_Float16*)(ws + off); off += WT_SZ;
    _Float16* Wot = (_Float16*)(ws + off); off += WOT_SZ;
    _Float16* Qp  = (_Float16*)(ws + off); off += P_SZ;
    _Float16* Kp  = (_Float16*)(ws + off); off += P_SZ;
    _Float16* Vt  = (_Float16*)(ws + off); off += P_SZ;
    _Float16* Oc  = (_Float16*)(ws + off); off += OC_SZ;

    prep_weights<<<256, 256, 0, stream>>>(Wq, Wk, Wv, Wo, Wqt, Wkt, Wvt, Wot);

    proj_kernel<<<dim3(S_ / 16, H_, 3), 32, 0, stream>>>(
        q, k, v, Wqt, Wkt, Wvt, bq, bk, bv, Qp, Kp, Vt);

    flash_kernel<<<dim3(S_ / (16 * QW_), H_), 32 * QW_, 0, stream>>>(Qp, Kp, Vt, Oc);

    outproj_kernel<<<dim3(S_ / 16, OUT_ / 64), 32, 0, stream>>>(Oc, Wot, bo, out);
}
